// MultiHeadAttention_21835613733517
// MI455X (gfx1250) — matrix operand order repaired and checked by reading, not compiled, not run on hardware
//
#include <hip/hip_runtime.h>

typedef __bf16 bf16;
typedef bf16 v16bf __attribute__((ext_vector_type(16)));
typedef bf16 v4bf  __attribute__((ext_vector_type(4)));
typedef float v8f  __attribute__((ext_vector_type(8)));

#define D_MODEL 1024
#define N_HEADS 16
#define DK      64
#define SEQ     2048
#define BATCH   2
#define MTOT    (BATCH*SEQ)

union ABFrag { v16bf v; uint4 u[2]; };

__device__ __forceinline__ v8f wmma_bf16(const ABFrag& a, const ABFrag& b, v8f c) {
    // D = A(16x32 bf16) * B(32x16 bf16) + C(16x16 f32)
    return __builtin_amdgcn_wmma_f32_16x16x32_bf16(false, a.v, false, b.v,
                                                   (short)0, c, false, false);
}

// Async DMA: 16 bytes global -> LDS, tracked with ASYNCcnt (CDNA5 §15.18).
__device__ __forceinline__ void async_b128(void* lds, const void* g) {
    uint32_t l = (uint32_t)(uintptr_t)lds;      // low 32 bits = LDS byte offset
    uint64_t a = (uint64_t)(uintptr_t)g;
    asm volatile("global_load_async_to_lds_b128 %0, %1, off"
                 :: "v"(l), "v"(a) : "memory");
}
__device__ __forceinline__ void wait_async0() {
    asm volatile("s_wait_asynccnt 0x0" ::: "memory");
}

// ---------------------------------------------------------------------------
// f32 -> bf16 bulk conversion (4-wide)
// ---------------------------------------------------------------------------
__global__ __launch_bounds__(256)
void f32_to_bf16_kernel(const float* __restrict__ src, bf16* __restrict__ dst, int n4)
{
    int i = blockIdx.x * blockDim.x + threadIdx.x;
    if (i < n4) {
        float4 f = ((const float4*)src)[i];
        v4bf o;
        o[0] = (bf16)f.x; o[1] = (bf16)f.y; o[2] = (bf16)f.z; o[3] = (bf16)f.w;
        ((v4bf*)dst)[i] = o;
    }
}

// ---------------------------------------------------------------------------
// Wt[d][e] = (bf16) W[e][d]   (1024x1024), LDS tiled transpose
// ---------------------------------------------------------------------------
__global__ __launch_bounds__(256)
void transpose_convert_kernel(const float* __restrict__ W, bf16* __restrict__ Wt)
{
    __shared__ float tile[32][33];
    const int tx = threadIdx.x;   // 0..31
    const int ty = threadIdx.y;   // 0..7
    const int e0 = blockIdx.y * 32;
    const int d0 = blockIdx.x * 32;
    #pragma unroll
    for (int j = 0; j < 32; j += 8)
        tile[ty + j][tx] = W[(size_t)(e0 + ty + j) * D_MODEL + d0 + tx];
    __syncthreads();
    #pragma unroll
    for (int j = 0; j < 32; j += 8)
        Wt[(size_t)(d0 + ty + j) * D_MODEL + e0 + tx] = (bf16)tile[tx][ty + j];
}

// ---------------------------------------------------------------------------
// Tiled bf16 GEMM: Y(M x 1024) = X(M x 1024) @ Wt(1024 x 1024) + bias
// 256 threads = 8 waves; tile 128x64, BK=32; wave grid 4x2, 2x2 WMMA per wave.
// Double-buffered LDS filled by async global->LDS DMA; loop unrolled x2 so
// buffer indices are compile-time constants (no phi copies / buffer branch).
// mode 0: Q -> (B,H,S,Dk) bf16, *1/sqrt(Dk);  mode 1: K -> (B,H,Dk,S) bf16
// mode 2: V -> (B,H,S,Dk) bf16;               mode 3: f32 row-major out
// ---------------------------------------------------------------------------
__global__ __launch_bounds__(256)
void gemm_bf16_kernel(const bf16* __restrict__ X, const bf16* __restrict__ Wt,
                      const float* __restrict__ bias,
                      bf16* __restrict__ outB, float* __restrict__ outF, int mode)
{
    __shared__ alignas(16) bf16 As[2][128][40];  // 128x32 + pad, 16B rows
    __shared__ alignas(16) bf16 Bs[2][32][72];   // 32x64 + pad

    const int tid  = threadIdx.x;
    const int lane = tid & 31;
    const int w    = tid >> 5;
    const int wr   = w >> 1;          // 0..3
    const int wc   = w & 1;           // 0..1
    const int m0   = blockIdx.y * 128;
    const int n0   = blockIdx.x * 64;

    const int arow = tid >> 1;        // 0..127
    const int acol = (tid & 1) * 16;
    const int brow = tid >> 3;        // 0..31
    const int bcol = (tid & 7) * 8;
    const int r    = lane & 15;
    const int kb   = lane >> 4;       // selects K-halves per 16-bit A layout

    v8f zero = {0.f,0.f,0.f,0.f,0.f,0.f,0.f,0.f};
    v8f acc[4];
    #pragma unroll
    for (int i = 0; i < 4; ++i) acc[i] = zero;

    auto stage = [&](int ki, int buf) {
        const bf16* xs = X + (size_t)(m0 + arow) * D_MODEL + ki * 32 + acol;
        async_b128(&As[buf][arow][acol],     xs);
        async_b128(&As[buf][arow][acol + 8], xs + 8);
        async_b128(&Bs[buf][brow][bcol],
                   Wt + (size_t)(ki * 32 + brow) * D_MODEL + n0 + bcol);
    };

    auto computeTile = [&](int buf) {
        ABFrag a[2], bb[2];
        #pragma unroll
        for (int ms = 0; ms < 2; ++ms) {
            const uint4* p = (const uint4*)&As[buf][wr*32 + ms*16 + r][0];
            a[ms].u[0] = p[kb];        // K {0..7} or {8..15}
            a[ms].u[1] = p[2 + kb];    // K {16..23} or {24..31}
        }
        #pragma unroll
        for (int ns = 0; ns < 2; ++ns) {
            #pragma unroll
            for (int s = 0; s < 16; ++s)
                bb[ns].v[s] = Bs[buf][(s < 8) ? (8*kb + s) : (16 + 8*kb + s - 8)][wc*32 + ns*16 + r];
        }
        #pragma unroll
        for (int ms = 0; ms < 2; ++ms)
            #pragma unroll
            for (int ns = 0; ns < 2; ++ns)
                acc[ms*2 + ns] = wmma_bf16(a[ms], bb[ns], acc[ms*2 + ns]);
    };

    const int nK = D_MODEL / 32;       // 32 (even)
    stage(0, 0);                       // prologue fill
    for (int ki = 0; ki < nK; ki += 2) {
        wait_async0();                 // buf0 DMA landed (mine)
        __syncthreads();               // everyone's landed
        stage(ki + 1, 1);              // prefetch odd tile (ki+1 < nK always)
        computeTile(0);

        wait_async0();                 // buf1 DMA landed
        __syncthreads();
        if (ki + 2 < nK) stage(ki + 2, 0);
        computeTile(1);
    }

    // epilogue: C layout = lane -> N col, element -> M row
    const int ncol  = lane & 15;
    const int mbase = (lane >> 4) * 8;
    #pragma unroll
    for (int ms = 0; ms < 2; ++ms) {
        #pragma unroll
        for (int ns = 0; ns < 2; ++ns) {
            #pragma unroll
            for (int v = 0; v < 8; ++v) {
                const int gm = m0 + wr*32 + ms*16 + mbase + v;
                const int ge = n0 + wc*32 + ns*16 + ncol;
                const float val = acc[ms*2 + ns][v] + bias[ge];
                const int b = gm >> 11;          // / SEQ
                const int s = gm & (SEQ - 1);
                const int h = ge >> 6;           // / DK
                const int d = ge & (DK - 1);
                if (mode == 0) {
                    outB[(((size_t)(b*N_HEADS + h))*SEQ + s)*DK + d] = (bf16)(val * 0.125f);
                } else if (mode == 1) {
                    outB[(((size_t)(b*N_HEADS + h))*DK + d)*SEQ + s] = (bf16)val;
                } else if (mode == 2) {
                    outB[(((size_t)(b*N_HEADS + h))*SEQ + s)*DK + d] = (bf16)val;
                } else {
                    outF[(size_t)gm * D_MODEL + ge] = val;
                }
            }
        }
    }
}

// ---------------------------------------------------------------------------
// Flash attention per (b, h, 128-query block). 8 waves x 16 query rows.
// Q pre-scaled by 1/sqrt(Dk). K stored transposed (Dk,S); V stored (S,Dk).
// K/V chunks double-buffered via async DMA; loop unrolled x2.
// ---------------------------------------------------------------------------
__global__ __launch_bounds__(256)
void attention_kernel(const bf16* __restrict__ Q, const bf16* __restrict__ Kt,
                      const bf16* __restrict__ V, bf16* __restrict__ Ctx)
{
    __shared__ alignas(16) bf16 Ks[2][64][40];    // Kt chunk: 64d x 32keys + pad
    __shared__ alignas(16) bf16 Vs[2][32][72];    // V  chunk: 32keys x 64d + pad
    __shared__ alignas(16) bf16 Ps[8][16][40];    // per-wave P scratch 16x32 + pad

    const int tid  = threadIdx.x;
    const int lane = tid & 31;
    const int w    = tid >> 5;
    const int b    = blockIdx.z;
    const int h    = blockIdx.y;
    const int q0   = blockIdx.x * 128 + w * 16;

    const size_t bh = (size_t)(b * N_HEADS + h);
    const bf16* Qbh  = Q  + bh * SEQ * DK;
    const bf16* Ktbh = Kt + bh * DK * SEQ;
    const bf16* Vbh  = V  + bh * SEQ * DK;

    const int r     = lane & 15;
    const int kb    = lane >> 4;
    const int ncol  = lane & 15;
    const int mbase = (lane >> 4) * 8;

    const int krow = tid >> 2;         // 0..63
    const int kcol = (tid & 3) * 8;
    const int vrow = tid >> 3;         // 0..31
    const int vcol = (tid & 7) * 8;

    auto stageKV = [&](int kci, int buf) {
        async_b128(&Ks[buf][krow][kcol], Ktbh + (size_t)krow * SEQ + kci*32 + kcol);
        async_b128(&Vs[buf][vrow][vcol], Vbh + (size_t)(kci*32 + vrow) * DK + vcol);
    };

    // Q A-fragments live in registers for the whole pass
    ABFrag qa[2];
    #pragma unroll
    for (int dc = 0; dc < 2; ++dc) {
        const uint4* p = (const uint4*)(Qbh + (size_t)(q0 + r) * DK + dc*32);
        qa[dc].u[0] = p[kb];
        qa[dc].u[1] = p[2 + kb];
    }

    v8f zero = {0.f,0.f,0.f,0.f,0.f,0.f,0.f,0.f};
    v8f o[4];
    #pragma unroll
    for (int i = 0; i < 4; ++i) o[i] = zero;
    float m[8], l[8];
    #pragma unroll
    for (int v = 0; v < 8; ++v) { m[v] = -3.0e38f; l[v] = 0.f; }

    auto processChunk = [&](int buf) {
        // scores: S(16 x 32keys) = Q(16x64) . Kt(64x32), two 16x16 N-tiles
        v8f s0 = zero, s1 = zero;
        #pragma unroll
        for (int dc = 0; dc < 2; ++dc) {
            ABFrag kb0, kb1;
            #pragma unroll
            for (int s = 0; s < 16; ++s) {
                const int kd = dc*32 + ((s < 8) ? (8*kb + s) : (16 + 8*kb + s - 8));
                kb0.v[s] = Ks[buf][kd][r];        // keys 0..15
                kb1.v[s] = Ks[buf][kd][16 + r];   // keys 16..31
            }
            s0 = wmma_bf16(qa[dc], kb0, s0);
            s1 = wmma_bf16(qa[dc], kb1, s1);
        }

        // online softmax: row = C element; reduce across the 16-lane N group
        float alpha[8];
        #pragma unroll
        for (int v = 0; v < 8; ++v) {
            float t = fmaxf(s0[v], s1[v]);
            t = fmaxf(t, __shfl_xor(t, 1, 32));
            t = fmaxf(t, __shfl_xor(t, 2, 32));
            t = fmaxf(t, __shfl_xor(t, 4, 32));
            t = fmaxf(t, __shfl_xor(t, 8, 32));
            const float mn  = fmaxf(m[v], t);
            const float p0e = __expf(s0[v] - mn);
            const float p1e = __expf(s1[v] - mn);
            float rs = p0e + p1e;
            rs += __shfl_xor(rs, 1, 32);
            rs += __shfl_xor(rs, 2, 32);
            rs += __shfl_xor(rs, 4, 32);
            rs += __shfl_xor(rs, 8, 32);
            alpha[v] = __expf(m[v] - mn);
            l[v] = l[v] * alpha[v] + rs;
            m[v] = mn;
            s0[v] = p0e;
            s1[v] = p1e;
        }
        #pragma unroll
        for (int v = 0; v < 8; ++v) {
            o[0][v] *= alpha[v]; o[1][v] *= alpha[v];
            o[2][v] *= alpha[v]; o[3][v] *= alpha[v];
        }

        // C-layout -> A-layout reshape of P through per-wave LDS scratch
        #pragma unroll
        for (int v = 0; v < 8; ++v) {
            Ps[w][mbase + v][ncol]      = (bf16)s0[v];
            Ps[w][mbase + v][16 + ncol] = (bf16)s1[v];
        }
        asm volatile("s_wait_dscnt 0x0" ::: "memory");  // wave-local LDS RAW fence
        ABFrag pa;
        {
            const uint4* p = (const uint4*)&Ps[w][r][0];
            pa.u[0] = p[kb];
            pa.u[1] = p[2 + kb];
        }

        // O(16x64) += P(16x32) . V(32x64)
        #pragma unroll
        for (int nc = 0; nc < 4; ++nc) {
            ABFrag vb;
            #pragma unroll
            for (int s = 0; s < 16; ++s)
                vb.v[s] = Vs[buf][(s < 8) ? (8*kb + s) : (16 + 8*kb + s - 8)][nc*16 + r];
            o[nc] = wmma_bf16(pa, vb, o[nc]);
        }
    };

    const int nC = SEQ / 32;           // 64 (even)
    stageKV(0, 0);
    for (int kci = 0; kci < nC; kci += 2) {
        wait_async0();
        __syncthreads();
        stageKV(kci + 1, 1);           // kci+1 < nC always
        processChunk(0);

        wait_async0();
        __syncthreads();
        if (kci + 2 < nC) stageKV(kci + 2, 0);
        processChunk(1);
    }

    // epilogue: normalize and scatter to (B, S, D_MODEL) bf16 context
    float linv[8];
    #pragma unroll
    for (int v = 0; v < 8; ++v) linv[v] = 1.0f / l[v];
    #pragma unroll
    for (int nc = 0; nc < 4; ++nc) {
        #pragma unroll
        for (int v = 0; v < 8; ++v) {
            const int srow = q0 + mbase + v;
            const int dcol = nc*16 + ncol;
            Ctx[((size_t)(b*SEQ + srow)) * D_MODEL + h*DK + dcol] =
                (bf16)(o[nc][v] * linv[v]);
        }
    }
}

// ---------------------------------------------------------------------------
extern "C" void kernel_launch(void* const* d_in, const int* in_sizes, int n_in,
                              void* d_out, int out_size, void* d_ws, size_t ws_size,
                              hipStream_t stream)
{
    (void)in_sizes; (void)n_in; (void)out_size; (void)ws_size;
    const float* q  = (const float*)d_in[0];
    const float* k  = (const float*)d_in[1];
    const float* v  = (const float*)d_in[2];
    const float* wq = (const float*)d_in[3];
    const float* bq = (const float*)d_in[4];
    const float* wk = (const float*)d_in[5];
    const float* bk = (const float*)d_in[6];
    const float* wv = (const float*)d_in[7];
    const float* bv = (const float*)d_in[8];
    const float* wo = (const float*)d_in[9];
    const float* bo = (const float*)d_in[10];
    float* out = (float*)d_out;

    char* ws = (char*)d_ws;
    const size_t MB = 1ull << 20;
    bf16* Xq  = (bf16*)(ws +  0*MB);   // 4096x1024 bf16 (8 MiB)
    bf16* Xk  = (bf16*)(ws +  8*MB);
    bf16* Xv  = (bf16*)(ws + 16*MB);
    bf16* Wqt = (bf16*)(ws + 24*MB);   // 1024x1024 bf16 (2 MiB each)
    bf16* Wkt = (bf16*)(ws + 26*MB);
    bf16* Wvt = (bf16*)(ws + 28*MB);
    bf16* Wot = (bf16*)(ws + 30*MB);
    bf16* Qh  = (bf16*)(ws + 32*MB);   // (B,H,S,Dk)  8 MiB
    bf16* KtH = (bf16*)(ws + 40*MB);   // (B,H,Dk,S)  8 MiB
    bf16* Vh  = (bf16*)(ws + 48*MB);   // (B,H,S,Dk)  8 MiB
    bf16* Ctx = (bf16*)(ws + 56*MB);   // (B,S,D)     8 MiB

    const int n4 = MTOT * D_MODEL / 4;
    f32_to_bf16_kernel<<<(n4 + 255) / 256, 256, 0, stream>>>(q, Xq, n4);
    f32_to_bf16_kernel<<<(n4 + 255) / 256, 256, 0, stream>>>(k, Xk, n4);
    f32_to_bf16_kernel<<<(n4 + 255) / 256, 256, 0, stream>>>(v, Xv, n4);

    dim3 tb(32, 8), tg(32, 32);
    transpose_convert_kernel<<<tg, tb, 0, stream>>>(wq, Wqt);
    transpose_convert_kernel<<<tg, tb, 0, stream>>>(wk, Wkt);
    transpose_convert_kernel<<<tg, tb, 0, stream>>>(wv, Wvt);
    transpose_convert_kernel<<<tg, tb, 0, stream>>>(wo, Wot);

    dim3 gg(D_MODEL / 64, MTOT / 128);
    gemm_bf16_kernel<<<gg, 256, 0, stream>>>(Xq, Wqt, bq, Qh,  nullptr, 0);
    gemm_bf16_kernel<<<gg, 256, 0, stream>>>(Xk, Wkt, bk, KtH, nullptr, 1);
    gemm_bf16_kernel<<<gg, 256, 0, stream>>>(Xv, Wvt, bv, Vh,  nullptr, 2);

    dim3 ag(SEQ / 128, N_HEADS, BATCH);
    attention_kernel<<<ag, 256, 0, stream>>>(Qh, KtH, Vh, Ctx);

    gemm_bf16_kernel<<<gg, 256, 0, stream>>>(Ctx, Wot, bo, nullptr, out, 3);
}
